// MLP_73358041416164
// MI455X (gfx1250) — matrix operand order repaired and checked by reading, not compiled, not run on hardware
//
#include <hip/hip_runtime.h>
#include <hip/hip_bf16.h>
#include <math.h>

// ---------------------------------------------------------------------------
// Instant-NGP style NeRF forward, fused for gfx1250 (MI455X).
// Pass 1: convert/zero-pad all MLP weights to f16 in workspace (K padded to
//         multiples of 32): no guards / no converts in the hot loop.
// Pass 2: one wave32 per 32-point tile (two 16-row WMMA M tiles, so every
//         B fragment feeds two v_wmma_f32_16x16x32_f16 -> half the L2 weight
//         traffic). Hash-grid gather -> LDS activations -> 4 GEMMs -> fused
//         rgb head. Activations never touch HBM.
// ---------------------------------------------------------------------------

typedef __attribute__((ext_vector_type(16))) _Float16 v16h;
typedef __attribute__((ext_vector_type(8)))  float    v8f;

// grid metadata (compile-time, matches _grid_meta() in the reference)
__constant__ float g_scale[10] = {15.f,31.f,63.f,127.f,255.f,511.f,1023.f,2047.f,4095.f,8191.f};
__constant__ int   g_res1[10]  = {17,33,65,129,257,513,1025,2049,4097,8193};
__constant__ int   g_off[10]   = {0,4920,40864,315496,2412648,4509800,6606952,8704104,10801256,12898408};
// levels >= 3 are hashed; all hashed levels have 2^21 entries -> mask 0x1FFFFF

// f16 padded-weight layout inside d_ws (offsets in _Float16 elements)
#define W_D1P_OFF 0                       // 64  x 64
#define W_D2P_OFF 4096                    // 64  x 256
#define W_R0P_OFF 20480                   // 288 x 256
#define W_R1P_OFF 94208                   // 544 x 256
// total 233472 halfs = 466944 bytes of workspace

#define V8F_ZERO {0.f,0.f,0.f,0.f,0.f,0.f,0.f,0.f}

// ---------------- prep: f32 -> f16 with zero K-padding ---------------------
__global__ void cvt_pad_kernel(const float* __restrict__ src, _Float16* __restrict__ dst,
                               int krows, int kpad, int n) {
  int idx = blockIdx.x * blockDim.x + threadIdx.x;
  if (idx >= kpad * n) return;
  int row = idx / n;
  int col = idx - row * n;
  float v = (row < krows) ? src[(long)row * n + col] : 0.f;
  dst[idx] = (_Float16)v;
}

// A fragment (16x32 f16, M x K): lane = M (both halves), VGPR pairs = K:
//   v<4 : k = 8*half + 2v + {0,1} ; v>=4 : k = 16 + 8*half + 2(v-4) + {0,1}
// The 8 dword loads form two contiguous 16B runs -> 2x ds_load_b128.
static __device__ __forceinline__ v16h frag_a_lds(const _Float16* row, int k0, int half) {
  union { v16h v; unsigned u[8]; } f;
#pragma unroll
  for (int r = 0; r < 8; ++r) {
    int k = k0 + ((r < 4) ? (half * 8 + r * 2) : (16 + half * 8 + (r - 4) * 2));
    f.u[r] = *(const unsigned*)(row + k);   // k is even -> 4B aligned
  }
  return f.v;
}

// B fragment (32x16 f16, K x N) from pre-converted padded f16 weights:
// lane = N column (both halves), element s = K in table order (same as A).
// Rows beyond the true K are pre-zeroed.
static __device__ __forceinline__ v16h frag_b_ws(const _Float16* __restrict__ Wp,
                                                 int ldn, int k0, int n0, int lane) {
  union { v16h v; _Float16 h[16]; } f;
  const int hb = lane >> 4;
  const int n = n0 + (lane & 15);
#pragma unroll
  for (int s = 0; s < 16; ++s) {
    int k = k0 + ((s < 8) ? (hb * 8 + s) : (16 + hb * 8 + (s - 8)));
    f.h[s] = Wp[(long)k * ldn + n];
  }
  return f.v;
}

#define WMMA_F16(a, b, c) \
  __builtin_amdgcn_wmma_f32_16x16x32_f16(false, (a), false, (b), (short)0, (c), false, false)

__global__ __launch_bounds__(32) void nerf_fused_kernel(
    const float* __restrict__ coords, const float* __restrict__ viewdirs,
    const float* __restrict__ emb,
    const _Float16* __restrict__ wp,          // padded f16 weights in workspace
    const float* __restrict__ b_d1, const float* __restrict__ b_d2,
    const float* __restrict__ b_r0, const float* __restrict__ b_r1,
    const float* __restrict__ w_rgb, const float* __restrict__ b_rgb,
    float* __restrict__ out_cw, float* __restrict__ out_dens,
    float* __restrict__ out_rgb, int samples)
{
  __shared__ _Float16 s_feats[32 * 64];    // grid features, K padded 40->64
  __shared__ _Float16 s_h[32 * 64];        // d1 output
  __shared__ _Float16 s_inp[32 * 288];     // bottleneck(256) + dir_enc(27) + pad(5)
  __shared__ _Float16 s_y[32 * 256];       // r0 output      (total 43008 B)

  const _Float16* w_d1p = wp + W_D1P_OFF;
  const _Float16* w_d2p = wp + W_D2P_OFF;
  const _Float16* w_r0p = wp + W_R0P_OFF;
  const _Float16* w_r1p = wp + W_R1P_OFF;

  const int lane = threadIdx.x;            // 0..31 (one full wave32)
  const int ml   = lane & 15;              // M row within a 16-row tile == N lane
  const int half = lane >> 4;
  const int p0   = blockIdx.x * 32;
  const int ray  = p0 / samples;           // 32-point tile == exactly one ray

  // ---------------- phase 1: view-direction encoding -> s_inp[:,256:288] ----
  {
    _Float16 hv = (_Float16)0.f;           // lanes 27..31 write the K padding
    if (lane < 27) {
      float v;
      if (lane < 3) {
        v = viewdirs[ray * 3 + lane];
      } else {
        int t = lane - 3;
        int isc = (t >= 12) ? 1 : 0;
        t -= isc * 12;
        int s = t / 3, j = t - s * 3;
        float d  = viewdirs[ray * 3 + j];
        float xb = d * (float)(1 << s);
        v = isc ? cosf(xb) : sinf(xb);     // sin(xb + pi/2) == cos(xb)
      }
      hv = (_Float16)v;
    }
#pragma unroll 8
    for (int mm = 0; mm < 32; ++mm) s_inp[mm * 288 + 256 + lane] = hv;
  }

  // ---------------- phase 2: contract + hash-grid encode (lane == point) ----
  {
    const int p = p0 + lane;
    float cx = coords[p * 3 + 0], cy = coords[p * 3 + 1], cz = coords[p * 3 + 2];
    float m2 = fmaxf(1.1920929e-7f, cx * cx + cy * cy + cz * cz);
    if (m2 > 1.f) {
      float inv = (2.f * sqrtf(m2) - 1.f) / m2;
      cx *= inv; cy *= inv; cz *= inv;
    }
    float cwx = cx * 0.5f, cwy = cy * 0.5f, cwz = cz * 0.5f;   // contract / BOUND
    out_cw[p * 3 + 0] = cwx; out_cw[p * 3 + 1] = cwy; out_cw[p * 3 + 2] = cwz;
    float x0 = (cwx + 2.f) * 0.25f, x1 = (cwy + 2.f) * 0.25f, x2 = (cwz + 2.f) * 0.25f;

    // all 10 levels per lane; level loop is wave-uniform (scalar branch on l)
    for (int l = 0; l < 10; ++l) {
      float sc = g_scale[l];
      float px = x0 * sc + 0.5f, py = x1 * sc + 0.5f, pz = x2 * sc + 0.5f;
      float fx = floorf(px), fy = floorf(py), fz = floorf(pz);
      float rx = px - fx, ry = py - fy, rz = pz - fz;
      int   gx = (int)fx, gy = (int)fy, gz = (int)fz;
      int   res1 = g_res1[l];
      long  base = (long)g_off[l];
      float a0 = 0.f, a1 = 0.f, a2 = 0.f, a3 = 0.f;
#pragma unroll
      for (int c = 0; c < 8; ++c) {
        int ox = c & 1, oy = (c >> 1) & 1, oz = (c >> 2) & 1;
        unsigned ix = (unsigned)(gx + ox), iy = (unsigned)(gy + oy), iz = (unsigned)(gz + oz);
        unsigned idx;
        if (l >= 3)
          idx = (ix ^ (iy * 2654435761u) ^ (iz * 805459861u)) & 0x1FFFFFu;
        else
          idx = ix + iy * (unsigned)res1 + iz * (unsigned)(res1 * res1);
        float w = (ox ? rx : 1.f - rx) * (oy ? ry : 1.f - ry) * (oz ? rz : 1.f - rz);
        const float4 e = *(const float4*)(emb + (base + (long)idx) * 4);
        a0 += w * e.x; a1 += w * e.y; a2 += w * e.z; a3 += w * e.w;
      }
      s_feats[lane * 64 + l * 4 + 0] = (_Float16)a0;
      s_feats[lane * 64 + l * 4 + 1] = (_Float16)a1;
      s_feats[lane * 64 + l * 4 + 2] = (_Float16)a2;
      s_feats[lane * 64 + l * 4 + 3] = (_Float16)a3;
    }
#pragma unroll
    for (int j = 40; j < 64; ++j) s_feats[lane * 64 + j] = (_Float16)0.f;
  }
  __syncthreads();

  // ---------------- d1: feats[32x64(pad40)] @ w_d1 + b, relu -> s_h ---------
  {
    v16h a00 = frag_a_lds(s_feats + ml * 64,        0,  half);
    v16h a01 = frag_a_lds(s_feats + ml * 64,        32, half);
    v16h a10 = frag_a_lds(s_feats + (ml + 16) * 64, 0,  half);
    v16h a11 = frag_a_lds(s_feats + (ml + 16) * 64, 32, half);
    for (int nt = 0; nt < 4; ++nt) {
      v16h b0 = frag_b_ws(w_d1p, 64, 0,  nt * 16, lane);
      v16h b1 = frag_b_ws(w_d1p, 64, 32, nt * 16, lane);
      v8f acc0 = V8F_ZERO, acc1 = V8F_ZERO;
      acc0 = WMMA_F16(a00, b0, acc0);
      acc1 = WMMA_F16(a10, b0, acc1);
      acc0 = WMMA_F16(a01, b1, acc0);
      acc1 = WMMA_F16(a11, b1, acc1);
      int n = nt * 16 + ml;
      float bias = b_d1[n];
#pragma unroll
      for (int r = 0; r < 8; ++r) {
        s_h[(r + 8 * half) * 64 + n]        = (_Float16)fmaxf(acc0[r] + bias, 0.f);
        s_h[(r + 8 * half + 16) * 64 + n]   = (_Float16)fmaxf(acc1[r] + bias, 0.f);
      }
    }
  }
  __syncthreads();

  // ---------------- d2: h @ w_d2 + b -> density + bottleneck ----------------
  {
    v16h a00 = frag_a_lds(s_h + ml * 64,        0,  half);
    v16h a01 = frag_a_lds(s_h + ml * 64,        32, half);
    v16h a10 = frag_a_lds(s_h + (ml + 16) * 64, 0,  half);
    v16h a11 = frag_a_lds(s_h + (ml + 16) * 64, 32, half);
    for (int nt = 0; nt < 16; ++nt) {
      v16h b0 = frag_b_ws(w_d2p, 256, 0,  nt * 16, lane);
      v16h b1 = frag_b_ws(w_d2p, 256, 32, nt * 16, lane);
      v8f acc0 = V8F_ZERO, acc1 = V8F_ZERO;
      acc0 = WMMA_F16(a00, b0, acc0);
      acc1 = WMMA_F16(a10, b0, acc1);
      acc0 = WMMA_F16(a01, b1, acc0);
      acc1 = WMMA_F16(a11, b1, acc1);
      int n = nt * 16 + ml;
      float bias = b_d2[n];
#pragma unroll
      for (int r = 0; r < 8; ++r) {
        float v0 = acc0[r] + bias;         // no activation on d2
        float v1 = acc1[r] + bias;
        s_inp[(r + 8 * half) * 288 + n]      = (_Float16)v0;
        s_inp[(r + 8 * half + 16) * 288 + n] = (_Float16)v1;
        if (nt == 0 && ml == 0) {          // column n == 0: raw density
          float z0 = v0 - 1.f, z1 = v1 - 1.f;                  // DENSITY_BIAS
          out_dens[p0 + r + 8 * half]      = (z0 > 20.f) ? z0 : log1pf(expf(z0));
          out_dens[p0 + r + 8 * half + 16] = (z1 > 20.f) ? z1 : log1pf(expf(z1));
        }
      }
    }
  }
  __syncthreads();

  // ---------------- r0: inp[32x288(283)] @ w_r0 + b, relu -> s_y ------------
  for (int nt = 0; nt < 16; ++nt) {
    v8f acc0 = V8F_ZERO, acc1 = V8F_ZERO;
    for (int k0 = 0; k0 < 288; k0 += 32) {
      v16h b  = frag_b_ws(w_r0p, 256, k0, nt * 16, lane);
      v16h a0 = frag_a_lds(s_inp + ml * 288,        k0, half);
      v16h a1 = frag_a_lds(s_inp + (ml + 16) * 288, k0, half);
      acc0 = WMMA_F16(a0, b, acc0);
      acc1 = WMMA_F16(a1, b, acc1);
    }
    int n = nt * 16 + ml;
    float bias = b_r0[n];
#pragma unroll
    for (int r = 0; r < 8; ++r) {
      s_y[(r + 8 * half) * 256 + n]      = (_Float16)fmaxf(acc0[r] + bias, 0.f);
      s_y[(r + 8 * half + 16) * 256 + n] = (_Float16)fmaxf(acc1[r] + bias, 0.f);
    }
  }
  __syncthreads();

  // ------- r1: [y(256) ; inp(283)] @ w_r1 + b, relu; fused rgb head ---------
  float rgbacc[48];                        // [mt(2)][r(8)][j(3)]
#pragma unroll
  for (int i = 0; i < 48; ++i) rgbacc[i] = 0.f;

  for (int nt = 0; nt < 16; ++nt) {
    v8f acc0 = V8F_ZERO, acc1 = V8F_ZERO;
    for (int t = 0; t < 17; ++t) {         // 8 tiles from y, 9 tiles from inp
      v16h a0, a1, b;
      if (t < 8) {
        b  = frag_b_ws(w_r1p, 256, t * 32, nt * 16, lane);
        a0 = frag_a_lds(s_y + ml * 256,        t * 32, half);
        a1 = frag_a_lds(s_y + (ml + 16) * 256, t * 32, half);
      } else {
        b  = frag_b_ws(w_r1p, 256, 256 + (t - 8) * 32, nt * 16, lane);
        a0 = frag_a_lds(s_inp + ml * 288,        (t - 8) * 32, half);
        a1 = frag_a_lds(s_inp + (ml + 16) * 288, (t - 8) * 32, half);
      }
      acc0 = WMMA_F16(a0, b, acc0);
      acc1 = WMMA_F16(a1, b, acc1);
    }
    int n = nt * 16 + ml;
    float bias = b_r1[n];
    float wr0 = w_rgb[n * 3 + 0], wr1 = w_rgb[n * 3 + 1], wr2 = w_rgb[n * 3 + 2];
#pragma unroll
    for (int r = 0; r < 8; ++r) {
      float v0 = fmaxf(acc0[r] + bias, 0.f);
      float v1 = fmaxf(acc1[r] + bias, 0.f);
      rgbacc[r * 3 + 0]      += v0 * wr0;
      rgbacc[r * 3 + 1]      += v0 * wr1;
      rgbacc[r * 3 + 2]      += v0 * wr2;
      rgbacc[24 + r * 3 + 0] += v1 * wr0;
      rgbacc[24 + r * 3 + 1] += v1 * wr1;
      rgbacc[24 + r * 3 + 2] += v1 * wr2;
    }
  }
  // reduce partial rgb over the 16 N-lanes of each half (xor masks stay in-group)
#pragma unroll
  for (int i = 0; i < 48; ++i) {
    float v = rgbacc[i];
    v += __shfl_xor(v, 1, 32);
    v += __shfl_xor(v, 2, 32);
    v += __shfl_xor(v, 4, 32);
    v += __shfl_xor(v, 8, 32);
    rgbacc[i] = v;
  }
#pragma unroll
  for (int r = 0; r < 8; ++r) {
    if (ml == r) {                         // lane r of each half writes rows r+8h(+16)
#pragma unroll
      for (int mt = 0; mt < 2; ++mt) {
        int p = p0 + r + 8 * half + 16 * mt;
#pragma unroll
        for (int j = 0; j < 3; ++j) {
          float z = rgbacc[mt * 24 + r * 3 + j] + b_rgb[j];
          float s = 1.f / (1.f + expf(-z));
          out_rgb[p * 3 + j] = s * 1.002f - 0.001f;   // *(1+2*pad) - pad
        }
      }
    }
  }
}

extern "C" void kernel_launch(void* const* d_in, const int* in_sizes, int n_in,
                              void* d_out, int out_size, void* d_ws, size_t ws_size,
                              hipStream_t stream) {
  (void)n_in; (void)out_size; (void)ws_size;

  const float* coords   = (const float*)d_in[0];
  const float* viewdirs = (const float*)d_in[1];
  const float* emb      = (const float*)d_in[2];
  const float* w_d1  = (const float*)d_in[3];
  const float* b_d1  = (const float*)d_in[4];
  const float* w_d2  = (const float*)d_in[5];
  const float* b_d2  = (const float*)d_in[6];
  const float* w_r0  = (const float*)d_in[7];
  const float* b_r0  = (const float*)d_in[8];
  const float* w_r1  = (const float*)d_in[9];
  const float* b_r1  = (const float*)d_in[10];
  const float* w_rgb = (const float*)d_in[11];
  const float* b_rgb = (const float*)d_in[12];

  _Float16* wp = (_Float16*)d_ws;          // 466944 B of padded f16 weights

  // prep pass: convert + zero-pad each weight matrix (K -> multiple of 32)
  {
    int t;
    t = 64 * 64;
    cvt_pad_kernel<<<(t + 255) / 256, 256, 0, stream>>>(w_d1, wp + W_D1P_OFF, 40, 64, 64);
    t = 64 * 256;
    cvt_pad_kernel<<<(t + 255) / 256, 256, 0, stream>>>(w_d2, wp + W_D2P_OFF, 64, 64, 256);
    t = 288 * 256;
    cvt_pad_kernel<<<(t + 255) / 256, 256, 0, stream>>>(w_r0, wp + W_R0P_OFF, 283, 288, 256);
    t = 544 * 256;
    cvt_pad_kernel<<<(t + 255) / 256, 256, 0, stream>>>(w_r1, wp + W_R1P_OFF, 539, 544, 256);
  }

  int npts    = in_sizes[0] / 3;           // 65536
  int nrays   = in_sizes[1] / 3;           // 2048
  int samples = npts / nrays;              // 32

  float* out      = (float*)d_out;         // (cw, density, rgb) concatenated flat
  float* out_cw   = out;
  float* out_dens = out + (size_t)npts * 3;
  float* out_rgb  = out + (size_t)npts * 4;

  int blocks = npts / 32;                  // one wave32 per 32-point tile
  nerf_fused_kernel<<<blocks, 32, 0, stream>>>(
      coords, viewdirs, emb, wp, b_d1, b_d2, b_r0, b_r1, w_rgb, b_rgb,
      out_cw, out_dens, out_rgb, samples);
}
